// AttentionPooling_16870631539210
// MI455X (gfx1250) — compile-verified, matrix operand order checked by reading, not run on hardware
//
#include <hip/hip_runtime.h>

// ---------------- problem constants (from reference) ----------------
#define NROWS 200000
#define DDIM  256
#define HDIM  128
#define NSEG  512

#define ROWS_PER_BLOCK 128          // 8 waves * 16 rows
#define XT_STRIDE      260          // padded f32 row stride in LDS (256 + 4 dw pad)

typedef __attribute__((ext_vector_type(16))) _Float16 v16h;
typedef __attribute__((ext_vector_type(8)))  float    v8f;
typedef __attribute__((ext_vector_type(4)))  float    v4f;
typedef __attribute__((ext_vector_type(4)))  unsigned v4u;
typedef __attribute__((ext_vector_type(8)))  int      v8i;
typedef __attribute__((ext_vector_type(4)))  int      v4i;

// TDM availability (this toolchain: 6-arg form, extra int32x8 group before cpol).
#if defined(__HIP_DEVICE_COMPILE__) && defined(__has_builtin)
#if __has_builtin(__builtin_amdgcn_tensor_load_to_lds) && __has_builtin(__builtin_amdgcn_s_wait_tensorcnt)
#define USE_TDM 1
#endif
#endif
#ifndef USE_TDM
#define USE_TDM 0
#endif

// order-preserving float<->uint map so atomicMax(u32) == float max
__device__ __forceinline__ unsigned ordEnc(float f) {
    unsigned u = __float_as_uint(f);
    return (u & 0x80000000u) ? ~u : (u | 0x80000000u);
}
__device__ __forceinline__ float ordDec(unsigned u) {
    unsigned b = (u & 0x80000000u) ? (u & 0x7FFFFFFFu) : ~u;
    return __uint_as_float(b);
}

// K index pattern of a 16-bit WMMA A/B fragment half `e` for lane `l`
// (v16h element e -> dword e>>1, half e&1; per ISA 16-bit A 16x32 layout)
__device__ __host__ __forceinline__ int fragK(int l, int e) {
    int kb = (l >> 4) * 8;
    int j  = e >> 1;
    return kb + ((j < 4) ? (2 * j) : (16 + 2 * (j - 4))) + (e & 1);
}

// ---------------- K0: zero output + segment accumulators ----------------
__global__ void init_ws(float* __restrict__ out, float* __restrict__ segSum,
                        unsigned* __restrict__ segMaxU) {
    int i = blockIdx.x * 256 + threadIdx.x;
    if (i < NSEG * DDIM) out[i] = 0.0f;
    if (i < NSEG) { segSum[i] = 0.0f; segMaxU[i] = 0u; }   // ordEnc: 0 acts as -inf
}

// ---------------- K0b: repack W1 -> per-lane f16 B fragments ----------------
// bpack[(nt*8+kk)*512 + lane*16 + e] = (f16) W1[kk*32 + fragK(lane,e)][nt*16 + (lane&15)]
__global__ void prepack_w1(const float* __restrict__ W1, _Float16* __restrict__ bpack) {
    int idx = blockIdx.x * 256 + threadIdx.x;          // 0 .. 64*512-1
    if (idx >= 64 * 512) return;
    int frag = idx >> 9;
    int rem  = idx & 511;
    int l    = rem >> 4;
    int e    = rem & 15;
    int nt = frag >> 3, kk = frag & 7;
    int k   = kk * 32 + fragK(l, e);
    int col = nt * 16 + (l & 15);
    bpack[idx] = (_Float16)W1[k * HDIM + col];
}

// ---------------- K1: scores = tanh(x@W1+b1)@W2+b2 ; segment max ----------------
__global__ void __launch_bounds__(256)
score_kernel(const float* __restrict__ x, const int* __restrict__ batch,
             const _Float16* __restrict__ bpack,
             const float* __restrict__ b1, const float* __restrict__ W2,
             const float* __restrict__ b2,
             float* __restrict__ scores, unsigned* __restrict__ segMaxU) {
    extern __shared__ float smem[];                    // 8 waves * 16 rows * 260 f32
    const int tid  = threadIdx.x;
    const int wave = tid >> 5;
    const int lane = tid & 31;
    const int lm   = lane & 15;
    const int hw   = lane >> 4;
    const int rowBase = blockIdx.x * ROWS_PER_BLOCK + wave * 16;
    if (rowBase >= NROWS) return;                      // wave-uniform exit, no barriers used

    float* xt = smem + wave * 16 * XT_STRIDE;

#if USE_TDM
    {   // 2D TDM descriptor: tile 256x16 f32, tensor rows = NROWS-rowBase (OOB -> zero-fill),
        // LDS pad: 4 dwords every 256 dwords -> row stride 260 f32 (bank-conflict-free reads).
        unsigned ldsAddr = (unsigned)(size_t)(void*)xt;
        unsigned long long ga = (unsigned long long)(size_t)(const void*)(x + (size_t)rowBase * DDIM);
        unsigned dim1 = (unsigned)(NROWS - rowBase);
        v4u g0;
        g0[0] = 1u;                                            // count=1, user descriptor
        g0[1] = ldsAddr;                                       // lds_addr
        g0[2] = (unsigned)(ga & 0xFFFFFFFFu);                  // global_addr[31:0]
        g0[3] = (unsigned)((ga >> 32) & 0x1FFFFFFu) | (2u << 30); // global_addr[56:32] | type=2
        v8i g1;
        g1[0] = (2 << 16) | (1 << 20) | (7 << 22) | (3 << 25); // data_size=4B, pad_en, 256dw interval, 4dw pad
        g1[1] = (int)(((unsigned)DDIM & 0xFFFFu) << 16);       // tensor_dim0 lo16
        g1[2] = (int)((((unsigned)DDIM >> 16) & 0xFFFFu) | ((dim1 & 0xFFFFu) << 16));
        g1[3] = (int)(((dim1 >> 16) & 0xFFFFu) | ((unsigned)DDIM << 16)); // tile_dim0=256
        g1[4] = 16;                                            // tile_dim1=16 rows, tile_dim2=0
        g1[5] = DDIM;                                          // tensor_dim0_stride lo32
        g1[6] = 0; g1[7] = 0;
        v4i g2 = {0, 0, 0, 0};
        v4i g3 = {0, 0, 0, 0};
        v8i g4 = {0, 0, 0, 0, 0, 0, 0, 0};                     // extra group on this toolchain
        __builtin_amdgcn_tensor_load_to_lds(g0, g1, g2, g3, g4, 0);
    }
#else
    for (int t = lane; t < 16 * DDIM; t += 32) {
        int r = t >> 8, c = t & (DDIM - 1);
        int row = rowBase + r;
        xt[r * XT_STRIDE + c] = (row < NROWS) ? x[(size_t)row * DDIM + c] : 0.0f;
    }
#endif

    v8f acc[8];
#pragma unroll
    for (int nt = 0; nt < 8; ++nt) acc[nt] = (v8f){0.f,0.f,0.f,0.f,0.f,0.f,0.f,0.f};

#if USE_TDM
    __builtin_amdgcn_s_wait_tensorcnt(0);              // only this wave's rows needed
#endif

    const float* xrow = xt + lm * XT_STRIDE;           // this lane's A row (M = lm)
#pragma unroll
    for (int kk = 0; kk < 8; ++kk) {
        int c0 = kk * 32 + hw * 8;
        v4f f0 = *(const v4f*)(xrow + c0);
        v4f f1 = *(const v4f*)(xrow + c0 + 4);
        v4f f2 = *(const v4f*)(xrow + c0 + 16);
        v4f f3 = *(const v4f*)(xrow + c0 + 20);
        v16h a;
        a[0]  = (_Float16)f0[0];  a[1]  = (_Float16)f0[1];
        a[2]  = (_Float16)f0[2];  a[3]  = (_Float16)f0[3];
        a[4]  = (_Float16)f1[0];  a[5]  = (_Float16)f1[1];
        a[6]  = (_Float16)f1[2];  a[7]  = (_Float16)f1[3];
        a[8]  = (_Float16)f2[0];  a[9]  = (_Float16)f2[1];
        a[10] = (_Float16)f2[2];  a[11] = (_Float16)f2[3];
        a[12] = (_Float16)f3[0];  a[13] = (_Float16)f3[1];
        a[14] = (_Float16)f3[2];  a[15] = (_Float16)f3[3];
#pragma unroll
        for (int nt = 0; nt < 8; ++nt) {
            v16h bfr = *(const v16h*)(bpack + ((nt * 8 + kk) * 512) + lane * 16);
            acc[nt] = __builtin_amdgcn_wmma_f32_16x16x32_f16(
                false, a, false, bfr, (short)0, acc[nt], false, false);
        }
    }

    // epilogue: tanh + second layer, fully in f32
    float p[8];
#pragma unroll
    for (int v = 0; v < 8; ++v) p[v] = 0.0f;
#pragma unroll
    for (int nt = 0; nt < 8; ++nt) {
        int col = nt * 16 + lm;
        float b1v = b1[col];
        float w2v = W2[col];
#pragma unroll
        for (int v = 0; v < 8; ++v)
            p[v] += tanhf(acc[nt][v] + b1v) * w2v;
    }
    // reduce across the 16 lanes holding this row's columns (xor<16 stays in half-wave)
#pragma unroll
    for (int m = 8; m >= 1; m >>= 1) {
#pragma unroll
        for (int v = 0; v < 8; ++v) p[v] += __shfl_xor(p[v], m, 32);
    }
    if (lm == 0) {
        float b2v = b2[0];
#pragma unroll
        for (int v = 0; v < 8; ++v) {
            int row = rowBase + hw * 8 + v;             // C layout: lanes>=16 hold M=v+8
            if (row < NROWS) {
                float s = p[v] + b2v;
                scores[row] = s;
                atomicMax(&segMaxU[batch[row]], ordEnc(s));
            }
        }
    }
}

// ---------------- K2: e = exp(s - segmax); segment sums (wave-coalesced) ----------------
__global__ void exp_kernel(const int* __restrict__ batch, float* __restrict__ sc,
                           const unsigned* __restrict__ segMaxU, float* __restrict__ segSum) {
    int i = blockIdx.x * 256 + threadIdx.x;
    bool valid = i < NROWS;
    int g = valid ? batch[i] : -1;
    float e = 0.0f;
    if (valid) {
        e = __expf(sc[i] - ordDec(segMaxU[g]));
        sc[i] = e;                                     // overwrite scores with e (same buffer)
    }
    int g0 = __shfl(g, 0, 32);
    if (__all(g == g0) && g0 >= 0) {                   // sorted batch -> usually uniform
        float s = e;
#pragma unroll
        for (int m = 16; m >= 1; m >>= 1) s += __shfl_xor(s, m, 32);
        if ((threadIdx.x & 31) == 0) atomicAdd(&segSum[g0], s);
    } else if (valid) {
        atomicAdd(&segSum[g], e);
    }
}

// ---------------- K3: out[g][:] += w_i * x[i][:] (sorted rows, boundary atomics) ----------------
__global__ void __launch_bounds__(256)
pool_kernel(const float* __restrict__ x, const int* __restrict__ batch,
            const float* __restrict__ eArr, const float* __restrict__ segSum,
            float* __restrict__ out) {
    __shared__ float wrow[256];
    __shared__ int   srow[256];
    int tid = threadIdx.x;
    int r0  = blockIdx.x * 256;
    int row = r0 + tid;
    if (row < NROWS) {
        int g = batch[row];
        float den = segSum[g];
        wrow[tid] = eArr[row] / (den > 0.0f ? den : 1.0f);
        srow[tid] = g;
    } else {
        wrow[tid] = 0.0f;
        srow[tid] = -1;
    }
    __syncthreads();

    int nr = NROWS - r0; if (nr > 256) nr = 256;
    float acc = 0.0f;
    int cur = srow[0];
    for (int r = 0; r < nr; ++r) {
        int g = srow[r];                               // LDS broadcast
        if (g != cur) {
            atomicAdd(&out[(size_t)cur * DDIM + tid], acc);
            acc = 0.0f;
            cur = g;
        }
        acc += x[((size_t)(r0 + r)) * DDIM + tid] * wrow[r];   // coalesced 1KB/row
    }
    if (cur >= 0) atomicAdd(&out[(size_t)cur * DDIM + tid], acc);
}

// ---------------- launcher ----------------
extern "C" void kernel_launch(void* const* d_in, const int* in_sizes, int n_in,
                              void* d_out, int out_size, void* d_ws, size_t ws_size,
                              hipStream_t stream) {
    const float* x     = (const float*)d_in[0];
    const int*   batch = (const int*)d_in[1];
    const float* W1    = (const float*)d_in[2];
    const float* b1    = (const float*)d_in[3];
    const float* W2    = (const float*)d_in[4];
    const float* b2    = (const float*)d_in[5];
    float* out = (float*)d_out;

    // workspace layout: scores/e [N] f32 | segSum [512] f32 | segMaxU [512] u32 | bpack [32768] f16
    float*     sc      = (float*)d_ws;
    float*     segSum  = sc + NROWS;
    unsigned*  segMaxU = (unsigned*)(segSum + NSEG);
    _Float16*  bpack   = (_Float16*)(segMaxU + NSEG);  // offset 804096 B: 32B-aligned for v16h loads

    init_ws<<<(NSEG * DDIM) / 256, 256, 0, stream>>>(out, segSum, segMaxU);
    prepack_w1<<<(64 * 512) / 256, 256, 0, stream>>>(W1, bpack);

    size_t ldsBytes = (size_t)ROWS_PER_BLOCK * XT_STRIDE * sizeof(float);  // 133120 B -> 2 WGs/WGP
    int nBlocks = (NROWS + ROWS_PER_BLOCK - 1) / ROWS_PER_BLOCK;           // 1563
    score_kernel<<<nBlocks, 256, ldsBytes, stream>>>(x, batch, bpack, b1, W2, b2, sc, segMaxU);

    int nElemBlocks = (NROWS + 255) / 256;                                  // 782
    exp_kernel<<<nElemBlocks, 256, 0, stream>>>(batch, sc, segMaxU, segSum);
    pool_kernel<<<nElemBlocks, 256, 0, stream>>>(x, batch, sc, segSum, out);
}
